// AdaptivePeriodicLayer_43791486550078
// MI455X (gfx1250) — compile-verified, matrix operand order checked by reading, not run on hardware
//
#include <hip/hip_runtime.h>
#include <math.h>

// ---------------- problem sizes ----------------
#define BB    8
#define SS    512
#define DD    128
#define DFFC  256
#define KTOP  3
#define NBK   (BB*KTOP)
#define NTAPS 35          // 1x1 + 3x3 + 5x5
#define LPMAX 1024        // max padded length (true max is 680)
#define NF    257         // rfft bins for S=512

// register blocking: each wave computes a (MT*16) x (NT*16) tile
#define MT 2
#define NT 4

typedef __attribute__((ext_vector_type(16))) _Float16 v16h;
typedef __attribute__((ext_vector_type(8)))  _Float16 v8h;
typedef __attribute__((ext_vector_type(8)))  float    v8f;

union Frag { v16h v; v8h h[2]; };

// tap -> spatial shift tables (t0 = 1x1, t1..9 = 3x3, t10..34 = 5x5)
__device__ __constant__ int TAP_DH[NTAPS] = {
  0,
  -1,-1,-1, 0,0,0, 1,1,1,
  -2,-2,-2,-2,-2, -1,-1,-1,-1,-1, 0,0,0,0,0, 1,1,1,1,1, 2,2,2,2,2
};
__device__ __constant__ int TAP_DW[NTAPS] = {
  0,
  -1,0,1, -1,0,1, -1,0,1,
  -2,-1,0,1,2, -2,-1,0,1,2, -2,-1,0,1,2, -2,-1,0,1,2, -2,-1,0,1,2
};

__device__ __forceinline__ float gelu_exact(float x) {
  return 0.5f * x * (1.0f + erff(x * 0.70710678118654752440f));
}

__device__ __forceinline__ void tap_decode_src(int t, int& ksel, int& ks, int& kh, int& kw) {
  if (t == 0)      { ksel = 0; ks = 1; kh = 0; kw = 0; }
  else if (t < 10) { ksel = 1; ks = 3; int i = t - 1;  kh = i / 3; kw = i % 3; }
  else             { ksel = 2; ks = 5; int i = t - 10; kh = i / 5; kw = i % 5; }
}

// ---------------- prep: trig tables + averaged biases + zero row ----------------
__global__ void prep_small(float* ct, float* st,
                           const float* b10, const float* b11, const float* b12,
                           const float* b20, const float* b21, const float* b22,
                           float* bias1, float* bias2, _Float16* zrow) {
  const int i = blockIdx.x * blockDim.x + threadIdx.x;
  if (i < SS) {
    const float ang = 6.28318530717958647692f * (float)i / (float)SS;
    ct[i] = cosf(ang);
    st[i] = sinf(ang);
  }
  if (i < DFFC) { bias1[i] = (b10[i] + b11[i] + b12[i]) * (1.0f / 3.0f); zrow[i] = (_Float16)0.0f; }
  if (i < DD)   bias2[i] = (b20[i] + b21[i] + b22[i]) * (1.0f / 3.0f);
}

// x (f32) -> f16
__global__ void cvt_x(const float* __restrict__ x, _Float16* __restrict__ xh) {
  const int n = BB * SS * DD;
  for (int j = blockIdx.x * blockDim.x + threadIdx.x; j < n; j += gridDim.x * blockDim.x)
    xh[j] = (_Float16)x[j];
}

// inception weights -> f16, layout [tap][cout][cin], scaled by 1/3 (branch average)
__global__ void cvt_conv_w(const float* __restrict__ w0, const float* __restrict__ w1,
                           const float* __restrict__ w2, _Float16* __restrict__ dst,
                           int cout, int cin) {
  const int n = NTAPS * cout * cin;
  for (int j = blockIdx.x * blockDim.x + threadIdx.x; j < n; j += gridDim.x * blockDim.x) {
    const int t   = j / (cout * cin);
    const int rem = j % (cout * cin);
    const int co = rem / cin, ci = rem % cin;
    int ksel, ks, kh, kw; tap_decode_src(t, ksel, ks, kh, kw);
    const float* w = (ksel == 0) ? w0 : ((ksel == 1) ? w1 : w2);
    const float v = w[(((size_t)co * cin + ci) * ks + kh) * ks + kw];
    dst[j] = (_Float16)(v * (1.0f / 3.0f));
  }
}

// enhancement weight [din][dout] -> f16 transposed [dout][din]
__global__ void cvt_enh(const float* __restrict__ ew, _Float16* __restrict__ ewh) {
  const int j = blockIdx.x * blockDim.x + threadIdx.x;
  if (j < DD * DD) {
    const int dout = j / DD, din = j % DD;
    ewh[j] = (_Float16)ew[(size_t)din * DD + dout];
  }
}

// ---------------- DFT magnitude spectrum: xf[b,f] = mean_d |rfft(x)[f,d]| ----------------
__global__ void dft_mag(const float* __restrict__ x, const float* __restrict__ ct,
                        const float* __restrict__ st, float* __restrict__ xf) {
  const int f = blockIdx.x;   // 0..256
  const int b = blockIdx.y;
  const int d = threadIdx.x;  // 0..127 (coalesced across d)
  const float* xb = x + (size_t)b * SS * DD + d;
  float re = 0.f, im = 0.f;
  int idx = 0;
  for (int s = 0; s < SS; ++s) {
    const float v = xb[(size_t)s * DD];
    re += v * ct[idx];
    im += v * st[idx];
    idx = (idx + f) & (SS - 1);
  }
  __shared__ float red[DD];
  red[d] = sqrtf(re * re + im * im);
  __syncthreads();
  for (int sft = DD / 2; sft > 0; sft >>= 1) {
    if (d < sft) red[d] += red[d + sft];
    __syncthreads();
  }
  if (d == 0) xf[(size_t)b * NF + f] = (f == 0) ? 0.f : red[0] * (1.0f / DD);
}

// ---------------- top-3 frequencies -> per-(b,k) geometry ----------------
__global__ void topk_meta(const float* __restrict__ xf, int* __restrict__ meta,
                          float* __restrict__ metaw) {
  const int b = threadIdx.x;
  if (b >= BB) return;
  const float* row = xf + (size_t)b * NF;
  int   bi0 = 0, bi1 = 0, bi2 = 0;
  float bv0 = -1e30f, bv1 = -1e30f, bv2 = -1e30f;
  for (int f = 0; f < NF; ++f) {
    const float v = row[f];
    if (v > bv0)      { bv2 = bv1; bi2 = bi1; bv1 = bv0; bi1 = bi0; bv0 = v; bi0 = f; }
    else if (v > bv1) { bv2 = bv1; bi2 = bi1; bv1 = v; bi1 = f; }
    else if (v > bv2) { bv2 = v; bi2 = f; }
  }
  const int   fs[KTOP] = { bi0, bi1, bi2 };
  const float vs[KTOP] = { bv0, bv1, bv2 };
  for (int k = 0; k < KTOP; ++k) {
    const int f = fs[k];
    int per = (f == 0) ? SS : (SS / f);
    if (per < 1) per = 1;
    const int nseg = (SS + per - 1) / per;
    const int lp   = nseg * per;
    const int bk = b * KTOP + k;
    meta[bk * 4 + 0] = f;
    meta[bk * 4 + 1] = per;
    meta[bk * 4 + 2] = nseg;
    meta[bk * 4 + 3] = lp;
    metaw[bk] = vs[k];
  }
}

// ---------------- conv1 (128 -> 256) implicit GEMM on WMMA, then GELU ----------------
// one wave computes a 32(pixels) x 64(cout) tile; K-loop = 35 taps x 128 cin.
// invalid (SAME-padding / tail) rows read a zero row via per-lane pointer select,
// keeping EXEC uniform through all loads + WMMAs.
__global__ void conv1_wmma(const _Float16* __restrict__ xh, const _Float16* __restrict__ w1h,
                           const float* __restrict__ bias1, const int* __restrict__ meta,
                           const _Float16* __restrict__ zrow, _Float16* __restrict__ h1) {
  const int bk = blockIdx.z;
  const int period = meta[bk * 4 + 1], nseg = meta[bk * 4 + 2], lp = meta[bk * 4 + 3];
  const int base = blockIdx.x * (MT * 16);
  if (base >= lp) return;
  const int ntb = blockIdx.y * (NT * 16);
  const int b = bk / KTOP;
  const int lane = threadIdx.x;
  const int m = lane & 15, lh = lane >> 4;
  const _Float16* xb = xh + (size_t)b * SS * DD;
  int pix[MT], h0[MT], w0[MT];
#pragma unroll
  for (int mi = 0; mi < MT; ++mi) {
    pix[mi] = base + mi * 16 + m;     // pixel flat index == sequence index
    h0[mi]  = pix[mi] % period;
    w0[mi]  = pix[mi] / period;
  }
  v8f acc[MT][NT] = {};
  for (int t = 0; t < NTAPS; ++t) {
    const int dh = TAP_DH[t], dw = TAP_DW[t];
    const _Float16* arow[MT];
#pragma unroll
    for (int mi = 0; mi < MT; ++mi) {
      const int hh = h0[mi] + dh, ww = w0[mi] + dw;
      const int sp = pix[mi] + dw * period + dh;
      const bool valid = (hh >= 0) & (hh < period) & (ww >= 0) & (ww < nseg) & (sp < SS);
      arow[mi] = valid ? (xb + (size_t)sp * DD) : zrow;
    }
    const _Float16* btap = w1h + (size_t)t * DFFC * DD;
#pragma unroll
    for (int cb = 0; cb < DD / 32; ++cb) {
      const int k0 = cb * 32 + lh * 8;
      Frag A[MT];
#pragma unroll
      for (int mi = 0; mi < MT; ++mi) {
        A[mi].h[0] = *(const v8h*)(arow[mi] + k0);
        A[mi].h[1] = *(const v8h*)(arow[mi] + k0 + 16);
      }
#pragma unroll
      for (int nt = 0; nt < NT; ++nt) {
        const _Float16* brow = btap + (size_t)(ntb + nt * 16 + m) * DD;
        Frag Bf;
        Bf.h[0] = *(const v8h*)(brow + k0);
        Bf.h[1] = *(const v8h*)(brow + k0 + 16);
#pragma unroll
        for (int mi = 0; mi < MT; ++mi)
          acc[mi][nt] = __builtin_amdgcn_wmma_f32_16x16x32_f16(false, A[mi].v, false, Bf.v,
                                                               (short)0, acc[mi][nt], false, false);
      }
    }
  }
  _Float16* outb = h1 + (size_t)bk * LPMAX * DFFC;
#pragma unroll
  for (int nt = 0; nt < NT; ++nt) {
    const int co = ntb + nt * 16 + m;
    const float bv = bias1[co];
#pragma unroll
    for (int mi = 0; mi < MT; ++mi) {
#pragma unroll
      for (int r = 0; r < 8; ++r) {
        const int pr = base + mi * 16 + lh * 8 + r;
        outb[(size_t)pr * DFFC + co] = (_Float16)gelu_exact(acc[mi][nt][r] + bv);
      }
    }
  }
}

// ---------------- conv2 (256 -> 128), only first S pixels needed downstream ----------------
__global__ void conv2_wmma(const _Float16* __restrict__ h1, const _Float16* __restrict__ w2h,
                           const float* __restrict__ bias2, const int* __restrict__ meta,
                           const _Float16* __restrict__ zrow, _Float16* __restrict__ h2) {
  const int bk = blockIdx.z;
  const int period = meta[bk * 4 + 1], nseg = meta[bk * 4 + 2];
  const int base = blockIdx.x * (MT * 16);   // 0..480 (lp >= 512 always)
  const int ntb = blockIdx.y * (NT * 16);
  const int lane = threadIdx.x;
  const int m = lane & 15, lh = lane >> 4;
  const _Float16* inb = h1 + (size_t)bk * LPMAX * DFFC;
  int pix[MT], h0[MT], w0[MT];
#pragma unroll
  for (int mi = 0; mi < MT; ++mi) {
    pix[mi] = base + mi * 16 + m;
    h0[mi]  = pix[mi] % period;
    w0[mi]  = pix[mi] / period;
  }
  v8f acc[MT][NT] = {};
  for (int t = 0; t < NTAPS; ++t) {
    const int dh = TAP_DH[t], dw = TAP_DW[t];
    const _Float16* arow[MT];
#pragma unroll
    for (int mi = 0; mi < MT; ++mi) {
      const int hh = h0[mi] + dh, ww = w0[mi] + dw;
      const int sp = pix[mi] + dw * period + dh;
      const bool valid = (hh >= 0) & (hh < period) & (ww >= 0) & (ww < nseg);
      arow[mi] = valid ? (inb + (size_t)sp * DFFC) : zrow;
    }
    const _Float16* btap = w2h + (size_t)t * DD * DFFC;
#pragma unroll
    for (int cb = 0; cb < DFFC / 32; ++cb) {
      const int k0 = cb * 32 + lh * 8;
      Frag A[MT];
#pragma unroll
      for (int mi = 0; mi < MT; ++mi) {
        A[mi].h[0] = *(const v8h*)(arow[mi] + k0);
        A[mi].h[1] = *(const v8h*)(arow[mi] + k0 + 16);
      }
#pragma unroll
      for (int nt = 0; nt < NT; ++nt) {
        const _Float16* brow = btap + (size_t)(ntb + nt * 16 + m) * DFFC;
        Frag Bf;
        Bf.h[0] = *(const v8h*)(brow + k0);
        Bf.h[1] = *(const v8h*)(brow + k0 + 16);
#pragma unroll
        for (int mi = 0; mi < MT; ++mi)
          acc[mi][nt] = __builtin_amdgcn_wmma_f32_16x16x32_f16(false, A[mi].v, false, Bf.v,
                                                               (short)0, acc[mi][nt], false, false);
      }
    }
  }
  _Float16* outb = h2 + (size_t)bk * SS * DD;
#pragma unroll
  for (int nt = 0; nt < NT; ++nt) {
    const int co = ntb + nt * 16 + m;
    const float bv = bias2[co];
#pragma unroll
    for (int mi = 0; mi < MT; ++mi) {
#pragma unroll
      for (int r = 0; r < 8; ++r) {
        const int pr = base + mi * 16 + lh * 8 + r;
        outb[(size_t)pr * DD + co] = (_Float16)(acc[mi][nt][r] + bv);
      }
    }
  }
}

// ---------------- enhancement GEMM (128x128) + GELU -> po (f32) ----------------
__global__ void enh_wmma(const _Float16* __restrict__ h2, const _Float16* __restrict__ ewh,
                         const float* __restrict__ eb, float* __restrict__ po) {
  const int bk = blockIdx.z;
  const int base = blockIdx.x * (MT * 16);   // 0..480
  const int ntb = blockIdx.y * (NT * 16);
  const int lane = threadIdx.x;
  const int m = lane & 15, lh = lane >> 4;
  const _Float16* inb = h2 + (size_t)bk * SS * DD;
  v8f acc[MT][NT] = {};
#pragma unroll
  for (int cb = 0; cb < DD / 32; ++cb) {
    const int k0 = cb * 32 + lh * 8;
    Frag A[MT];
#pragma unroll
    for (int mi = 0; mi < MT; ++mi) {
      const _Float16* arow = inb + (size_t)(base + mi * 16 + m) * DD;
      A[mi].h[0] = *(const v8h*)(arow + k0);
      A[mi].h[1] = *(const v8h*)(arow + k0 + 16);
    }
#pragma unroll
    for (int nt = 0; nt < NT; ++nt) {
      const _Float16* brow = ewh + (size_t)(ntb + nt * 16 + m) * DD;
      Frag Bf;
      Bf.h[0] = *(const v8h*)(brow + k0);
      Bf.h[1] = *(const v8h*)(brow + k0 + 16);
#pragma unroll
      for (int mi = 0; mi < MT; ++mi)
        acc[mi][nt] = __builtin_amdgcn_wmma_f32_16x16x32_f16(false, A[mi].v, false, Bf.v,
                                                             (short)0, acc[mi][nt], false, false);
    }
  }
  float* outb = po + ((size_t)bk << 16);     // SS*DD = 65536
#pragma unroll
  for (int nt = 0; nt < NT; ++nt) {
    const int co = ntb + nt * 16 + m;
    const float bv = eb[co];
#pragma unroll
    for (int mi = 0; mi < MT; ++mi) {
#pragma unroll
      for (int r = 0; r < 8; ++r) {
        const int pr = base + mi * 16 + lh * 8 + r;
        outb[(size_t)pr * DD + co] = gelu_exact(acc[mi][nt][r] + bv);
      }
    }
  }
}

// ---------------- attention blend weights cw[b,k] ----------------
__global__ void combine_weights(const float* __restrict__ po, const float* __restrict__ metaw,
                                const float* __restrict__ aw1, const float* __restrict__ ab1,
                                const float* __restrict__ aw2, const float* __restrict__ ab2,
                                float* __restrict__ cw) {
  const int b = blockIdx.x;
  const int t = threadIdx.x;  // 128
  __shared__ float red[DD];
  __shared__ float lsf[KTOP];
  for (int k = 0; k < KTOP; ++k) {
    red[t] = po[(((size_t)(b * KTOP + k)) * SS + (SS - 1)) * DD + t];
    __syncthreads();
    for (int s = DD / 2; s > 0; s >>= 1) {
      if (t < s) red[t] += red[t + s];
      __syncthreads();
    }
    if (t == 0) lsf[k] = red[0] * (1.0f / DD);
    __syncthreads();
  }
  if (t == 0) {
    float pw[KTOP], a[KTOP], z[KTOP];
    float mx = -1e30f;
    for (int k = 0; k < KTOP; ++k) mx = fmaxf(mx, metaw[b * KTOP + k]);
    float sm = 0.f;
    for (int k = 0; k < KTOP; ++k) { pw[k] = expf(metaw[b * KTOP + k] - mx); sm += pw[k]; }
    for (int k = 0; k < KTOP; ++k) pw[k] /= sm;
    float hid[2 * KTOP];
    for (int j = 0; j < 2 * KTOP; ++j) {
      float s2 = ab1[j];
      for (int k = 0; k < KTOP; ++k) s2 += lsf[k] * aw1[k * (2 * KTOP) + j];
      hid[j] = fmaxf(s2, 0.f);
    }
    for (int k = 0; k < KTOP; ++k) {
      float s2 = ab2[k];
      for (int j = 0; j < 2 * KTOP; ++j) s2 += hid[j] * aw2[j * KTOP + k];
      a[k] = s2;
    }
    mx = -1e30f;
    for (int k = 0; k < KTOP; ++k) mx = fmaxf(mx, a[k]);
    sm = 0.f;
    for (int k = 0; k < KTOP; ++k) { a[k] = expf(a[k] - mx); sm += a[k]; }
    for (int k = 0; k < KTOP; ++k) a[k] /= sm;
    mx = -1e30f;
    for (int k = 0; k < KTOP; ++k) { z[k] = pw[k] * a[k]; mx = fmaxf(mx, z[k]); }
    sm = 0.f;
    for (int k = 0; k < KTOP; ++k) { z[k] = expf(z[k] - mx); sm += z[k]; }
    for (int k = 0; k < KTOP; ++k) cw[b * KTOP + k] = z[k] / sm;
  }
}

// ---------------- out = x + sum_k po * cw ----------------
__global__ void finalize(const float* __restrict__ x, const float* __restrict__ po,
                         const float* __restrict__ cw, float* __restrict__ out) {
  const int n = BB * SS * DD;
  const int i = blockIdx.x * blockDim.x + threadIdx.x;
  if (i >= n) return;
  const int b = i >> 16;        // SS*DD = 65536
  const int sd = i & 65535;
  float acc = x[i];
  for (int k = 0; k < KTOP; ++k)
    acc += po[((size_t)(b * KTOP + k) << 16) + sd] * cw[b * KTOP + k];
  out[i] = acc;
}

// ---------------- launcher ----------------
extern "C" void kernel_launch(void* const* d_in, const int* in_sizes, int n_in,
                              void* d_out, int out_size, void* d_ws, size_t ws_size,
                              hipStream_t stream) {
  (void)in_sizes; (void)out_size; (void)ws_size;
  const float* x = (const float*)d_in[0];
  const float *c1w[3], *c1b[3], *c2w[3], *c2b[3];
  const float *aw1, *ab1, *aw2, *ab2, *ew, *eb;
  if (n_in >= 19) {  // lists flattened into separate inputs
    for (int i = 0; i < 3; ++i) {
      c1w[i] = (const float*)d_in[1 + i];
      c1b[i] = (const float*)d_in[4 + i];
      c2w[i] = (const float*)d_in[7 + i];
      c2b[i] = (const float*)d_in[10 + i];
    }
    aw1 = (const float*)d_in[13]; ab1 = (const float*)d_in[14];
    aw2 = (const float*)d_in[15]; ab2 = (const float*)d_in[16];
    ew  = (const float*)d_in[17]; eb  = (const float*)d_in[18];
  } else {           // lists concatenated per dict key (11 inputs)
    const float* w1c = (const float*)d_in[1];
    const float* b1c = (const float*)d_in[2];
    const float* w2c = (const float*)d_in[3];
    const float* b2c = (const float*)d_in[4];
    c1w[0] = w1c;            c1w[1] = w1c + 32768;  c1w[2] = w1c + 327680;
    c1b[0] = b1c;            c1b[1] = b1c + 256;    c1b[2] = b1c + 512;
    c2w[0] = w2c;            c2w[1] = w2c + 32768;  c2w[2] = w2c + 327680;
    c2b[0] = b2c;            c2b[1] = b2c + 128;    c2b[2] = b2c + 256;
    aw1 = (const float*)d_in[5]; ab1 = (const float*)d_in[6];
    aw2 = (const float*)d_in[7]; ab2 = (const float*)d_in[8];
    ew  = (const float*)d_in[9]; eb  = (const float*)d_in[10];
  }

  // workspace carve (~28 MB total)
  char* p = (char*)d_ws;
  auto carve = [&](size_t bytes) -> char* {
    char* r = p;
    p += (bytes + 255) & ~(size_t)255;
    return r;
  };
  float* ct    = (float*)carve(SS * 4);
  float* st    = (float*)carve(SS * 4);
  float* xf    = (float*)carve((size_t)BB * NF * 4);
  float* bias1 = (float*)carve(DFFC * 4);
  float* bias2 = (float*)carve(DD * 4);
  int*   meta  = (int*)carve((size_t)NBK * 4 * sizeof(int));
  float* metaw = (float*)carve(NBK * 4);
  float* cwb   = (float*)carve(NBK * 4);
  _Float16* zrow = (_Float16*)carve(DFFC * 2);
  _Float16* xh  = (_Float16*)carve((size_t)BB * SS * DD * 2);
  _Float16* w1h = (_Float16*)carve((size_t)NTAPS * DFFC * DD * 2);
  _Float16* w2h = (_Float16*)carve((size_t)NTAPS * DD * DFFC * 2);
  _Float16* ewh = (_Float16*)carve((size_t)DD * DD * 2);
  _Float16* h1  = (_Float16*)carve((size_t)NBK * LPMAX * DFFC * 2);
  _Float16* h2  = (_Float16*)carve((size_t)NBK * SS * DD * 2);
  float*    po  = (float*)carve((size_t)NBK * SS * DD * 4);

  // prep / conversions
  prep_small<<<2, 256, 0, stream>>>(ct, st, c1b[0], c1b[1], c1b[2],
                                    c2b[0], c2b[1], c2b[2], bias1, bias2, zrow);
  cvt_x<<<1024, 256, 0, stream>>>(x, xh);
  cvt_conv_w<<<2240, 256, 0, stream>>>(c1w[0], c1w[1], c1w[2], w1h, DFFC, DD);
  cvt_conv_w<<<2240, 256, 0, stream>>>(c2w[0], c2w[1], c2w[2], w2h, DD, DFFC);
  cvt_enh<<<64, 256, 0, stream>>>(ew, ewh);

  // spectrum + period selection (device-side; geometry is data-dependent)
  dft_mag<<<dim3(NF, BB), DD, 0, stream>>>(x, ct, st, xf);
  topk_meta<<<1, 32, 0, stream>>>(xf, meta, metaw);

  // heavy stage: inception convs + enhancement on WMMA (register-blocked 32x64 tiles)
  conv1_wmma<<<dim3(LPMAX / (MT * 16), DFFC / (NT * 16), NBK), 32, 0, stream>>>(
      xh, w1h, bias1, meta, zrow, h1);
  conv2_wmma<<<dim3(SS / (MT * 16), DD / (NT * 16), NBK), 32, 0, stream>>>(
      h1, w2h, bias2, meta, zrow, h2);
  enh_wmma<<<dim3(SS / (MT * 16), DD / (NT * 16), NBK), 32, 0, stream>>>(h2, ewh, eb, po);

  // blend + residual
  combine_weights<<<BB, DD, 0, stream>>>(po, metaw, aw1, ab1, aw2, ab2, cwb);
  finalize<<<(BB * SS * DD + 255) / 256, 256, 0, stream>>>(x, po, cwb, (float*)d_out);
}
